// CausalAttention_30107720745141
// MI455X (gfx1250) — matrix operand order repaired and checked by reading, not compiled, not run on hardware
//
#include <hip/hip_runtime.h>

// CDNA5 / gfx1250 causal attention:
//   kernel 1: QKV projection GEMM, 128x128 tile, double-buffered software
//             pipeline (global->regs->LDS overlapped with WMMA), fp32 -> bf16,
//             f32 accumulate; V stored transposed for the flash kernel
//   kernel 2: flash attention, Q staged in LDS, v_wmma_f32_16x16x32_bf16 for
//             QK^T and P@V, global_prefetch for next key block
// Workspace: 3 * B*S*D bf16 = 48 MiB (Q, K, V^T)

typedef __attribute__((ext_vector_type(16))) __bf16 bf16x16;
typedef __attribute__((ext_vector_type(8)))  float  floatx8;

#define S_LEN 2048
#define D_DIM 1024
#define BATCH 4

union FragBF {
  bf16x16 v;
  uint4   q[2];
};

__device__ __forceinline__ unsigned short f32_to_bf16(float f) {
  unsigned int u = __float_as_uint(f);
  unsigned int r = u + 0x7fffu + ((u >> 16) & 1u);   // round-to-nearest-even
  return (unsigned short)(r >> 16);
}
__device__ __forceinline__ unsigned int pack_bf16x2(float a, float b) {
  return (unsigned int)f32_to_bf16(a) | ((unsigned int)f32_to_bf16(b) << 16);
}
__device__ __forceinline__ floatx8 zero8() {
  floatx8 z;
#pragma unroll
  for (int i = 0; i < 8; ++i) z[i] = 0.0f;
  return z;
}

// ---------------------------------------------------------------------------
// Kernel 1: fused QKV projection. Grid (D/128, B*S/128, 3), block 256 (8 waves).
// Each wave computes a 16x128 strip (8 accumulator tiles of 16x16).
// Double-buffered: tile k+1 is loaded from global into registers while tile k
// is consumed by WMMAs from LDS; one barrier per K-step.
// ---------------------------------------------------------------------------
__global__ __launch_bounds__(256)
void qkv_proj_kernel(const float* __restrict__ x,
                     const float* __restrict__ Wq,
                     const float* __restrict__ Wk,
                     const float* __restrict__ Wv,
                     unsigned short* __restrict__ Q,
                     unsigned short* __restrict__ K,
                     unsigned short* __restrict__ Vt) {
  __shared__ __align__(16) unsigned short Xs[2][128][32];   // [m][k] bf16, 2x8 KB
  __shared__ __align__(16) unsigned short Ws[2][128][32];   // [n][k] bf16 (transposed), 2x8 KB

  const int t    = threadIdx.x;
  const int wave = t >> 5;
  const int lane = t & 31;
  const int half = lane >> 4;
  const int ln   = lane & 15;

  const int colBase = blockIdx.x * 128;
  const int rowBase = blockIdx.y * 128;
  const int which   = blockIdx.z;
  const float* W = (which == 0) ? Wq : (which == 1) ? Wk : Wv;

  floatx8 acc[8];
#pragma unroll
  for (int nt = 0; nt < 8; ++nt) acc[nt] = zero8();

  float4 xr[4], wr[4];

  auto load_tiles = [&](int k0) {
#pragma unroll
    for (int j = 0; j < 4; ++j) {
      int idx4 = t + j * 256;                    // 0..1023 float4s
      xr[j] = *(const float4*)(x + (size_t)(rowBase + (idx4 >> 3)) * D_DIM +
                               k0 + (idx4 & 7) * 4);
      wr[j] = *(const float4*)(W + (size_t)(k0 + (idx4 >> 5)) * D_DIM +
                               colBase + (idx4 & 31) * 4);
    }
  };
  auto store_tiles = [&](int bufi) {
#pragma unroll
    for (int j = 0; j < 4; ++j) {
      int idx4 = t + j * 256;
      // X: [row][k], packed pair stores (ds_store_b64)
      *(uint2*)(&Xs[bufi][idx4 >> 3][(idx4 & 7) * 4]) =
          make_uint2(pack_bf16x2(xr[j].x, xr[j].y), pack_bf16x2(xr[j].z, xr[j].w));
      // W: transposed [n][k]
      int kr = idx4 >> 5;
      int n4 = (idx4 & 31) * 4;
      Ws[bufi][n4 + 0][kr] = f32_to_bf16(wr[j].x);
      Ws[bufi][n4 + 1][kr] = f32_to_bf16(wr[j].y);
      Ws[bufi][n4 + 2][kr] = f32_to_bf16(wr[j].z);
      Ws[bufi][n4 + 3][kr] = f32_to_bf16(wr[j].w);
    }
  };

  // prologue: stage tile 0
  load_tiles(0);
  store_tiles(0);
  __syncthreads();

  for (int k0 = 0, it = 0; k0 < D_DIM; k0 += 32, ++it) {
    const int  cur  = it & 1;
    const bool more = (k0 + 32) < D_DIM;
    if (more) load_tiles(k0 + 32);   // global loads issue early, hide under WMMA

    // A fragment: row = wave*16 + ln; half-wave K split {0-15}/{16-31}, contiguous
    FragBF a;
    a.q[0] = *(const uint4*)(&Xs[cur][wave * 16 + ln][half * 16]);
    a.q[1] = *(const uint4*)(&Xs[cur][wave * 16 + ln][half * 16 + 8]);
#pragma unroll
    for (int nt = 0; nt < 8; ++nt) {
      // B fragment: col = nt*16 + ln; half-wave K split {0-15}/{16-31}, contiguous
      FragBF bfr;
      const unsigned short* bp = &Ws[cur][nt * 16 + ln][half * 16];
      bfr.q[0] = *(const uint4*)(bp);
      bfr.q[1] = *(const uint4*)(bp + 8);
      acc[nt] = __builtin_amdgcn_wmma_f32_16x16x32_bf16(
          false, a.v, false, bfr.v, (short)0, acc[nt], false, false);
    }

    if (more) store_tiles(cur ^ 1);  // fill the other buffer
    __syncthreads();
  }

  // epilogue: C layout -> lane 0-15: M=r, lane 16-31: M=8+r; N = ln
  // `which` is block-uniform: one branch, dedicated store loops.
  if (which == 2) {                                     // V, stored transposed [B][D][S]
#pragma unroll
    for (int nt = 0; nt < 8; ++nt) {
      int gcol = colBase + nt * 16 + ln;
#pragma unroll
      for (int r = 0; r < 8; ++r) {
        int grow = rowBase + wave * 16 + r + half * 8;
        int b = grow >> 11;
        int s = grow & (S_LEN - 1);
        Vt[(size_t)b * D_DIM * S_LEN + (size_t)gcol * S_LEN + s] = f32_to_bf16(acc[nt][r]);
      }
    }
  } else {
    unsigned short* __restrict__ dst = (which == 0) ? Q : K;
#pragma unroll
    for (int nt = 0; nt < 8; ++nt) {
      int gcol = colBase + nt * 16 + ln;
#pragma unroll
      for (int r = 0; r < 8; ++r) {
        int grow = rowBase + wave * 16 + r + half * 8;
        dst[(size_t)grow * D_DIM + gcol] = f32_to_bf16(acc[nt][r]);
      }
    }
  }
}

// ---------------------------------------------------------------------------
// Kernel 2: flash attention. Grid (B*S/16), block 256 (8 waves).
// Workgroup owns 16 queries (staged once into LDS); wave w owns output dims
// [w*128, w*128+128). Key blocks of 128: wave w computes scores for keys
// kBase+w*16..+15 over the full 1024-dim dot product.
// ---------------------------------------------------------------------------
__global__ __launch_bounds__(256)
void flash_attn_kernel(const unsigned short* __restrict__ Q,
                       const unsigned short* __restrict__ K,
                       const unsigned short* __restrict__ Vt,
                       float* __restrict__ out) {
  __shared__ __align__(16) unsigned short Qs[16][D_DIM];    // 32 KB, Q tile
  __shared__ __align__(16) float          scoresS[16][128]; // 8 KB
  __shared__ __align__(16) unsigned short Ps[16][128];      // 4 KB, bf16 P tile
  __shared__ float partmax[16][16];
  __shared__ float partsum[16][16];
  __shared__ float mrow[16];
  __shared__ float lrow[16];
  __shared__ float arow[16];
  __shared__ float mnewS[16];

  const int t    = threadIdx.x;
  const int wave = t >> 5;
  const int lane = t & 31;
  const int half = lane >> 4;
  const int ln   = lane & 15;

  const int qGlobal = blockIdx.x * 16;
  const int b       = qGlobal / S_LEN;
  const int qBase   = qGlobal & (S_LEN - 1);   // within batch

  const unsigned short* Qb = Q  + (size_t)b * S_LEN * D_DIM;
  const unsigned short* Kb = K  + (size_t)b * S_LEN * D_DIM;
  const unsigned short* Vb = Vt + (size_t)b * D_DIM * S_LEN;

  floatx8 accO[8];
#pragma unroll
  for (int i = 0; i < 8; ++i) accO[i] = zero8();

  // ---- stage the 16x1024 Q tile into LDS once (b128 loads)
#pragma unroll
  for (int j = 0; j < 8; ++j) {
    int idx = t + j * 256;         // 0..2047 uint4 chunks (8 bf16 each)
    int row = idx >> 7;            // 0..15
    int c8  = (idx & 127) * 8;     // 0..1016
    *(uint4*)(&Qs[row][c8]) = *(const uint4*)(Qb + (size_t)(qBase + row) * D_DIM + c8);
  }
  if (t < 16) { mrow[t] = -3.0e38f; lrow[t] = 0.0f; }
  __syncthreads();

  const float scale = 0.03125f;                // 1/sqrt(1024)
  const int   kEnd  = qBase + 15;              // last causally-visible key

  for (int kBase = 0; kBase <= kEnd; kBase += 128) {
    // prefetch next key block's K/V lines while we compute this one
    if (kBase + 128 <= kEnd) {
      __builtin_prefetch((const void*)(Kb + (size_t)(kBase + 128 + wave * 16 + ln) * D_DIM), 0, 1);
      __builtin_prefetch((const void*)(Vb + (size_t)(wave * 128 + ln * 8) * S_LEN + kBase + 128), 0, 1);
    }

    // ---- scores: S_tile[16q][16k] per wave over full K=1024 (32 WMMAs)
    const int keyCol = kBase + wave * 16 + ln;
    floatx8 sacc = zero8();
    const unsigned short* krow = Kb + (size_t)keyCol * D_DIM;
    for (int d0 = 0; d0 < D_DIM; d0 += 32) {
      FragBF a, bf;
      a.q[0]  = *(const uint4*)(&Qs[ln][d0 + half * 16]);
      a.q[1]  = *(const uint4*)(&Qs[ln][d0 + half * 16 + 8]);
      bf.q[0] = *(const uint4*)(krow + d0 + half * 16);
      bf.q[1] = *(const uint4*)(krow + d0 + half * 16 + 8);
      sacc = __builtin_amdgcn_wmma_f32_16x16x32_bf16(
          false, a.v, false, bf.v, (short)0, sacc, false, false);
    }
#pragma unroll
    for (int r = 0; r < 8; ++r)
      scoresS[r + half * 8][wave * 16 + ln] = sacc[r] * scale;
    __syncthreads();

    // ---- online softmax across the 128-key block (16 threads per query row)
    {
      const int row = t >> 4;
      const int sub = t & 15;
      float lmax = -3.0e38f;
#pragma unroll
      for (int j = 0; j < 8; ++j) {
        int col = sub * 8 + j;
        int key = kBase + col;
        float sv = (key <= qBase + row) ? scoresS[row][col] : -3.0e38f;  // causal
        scoresS[row][col] = sv;
        lmax = fmaxf(lmax, sv);
      }
      partmax[row][sub] = lmax;
      __syncthreads();
      float bmax = mrow[row];
#pragma unroll
      for (int j = 0; j < 16; ++j) bmax = fmaxf(bmax, partmax[row][j]);
      if (sub == 0) mnewS[row] = bmax;
      __syncthreads();
      float mnew = mnewS[row];
      float lsum = 0.0f;
#pragma unroll
      for (int j = 0; j < 8; ++j) {
        int col = sub * 8 + j;
        float p = __expf(scoresS[row][col] - mnew);
        Ps[row][col] = f32_to_bf16(p);
        lsum += p;
      }
      partsum[row][sub] = lsum;
      __syncthreads();
      if (sub == 0) {
        float alpha = __expf(mrow[row] - mnew);
        float ltot  = alpha * lrow[row];
#pragma unroll
        for (int j = 0; j < 16; ++j) ltot += partsum[row][j];
        arow[row] = alpha;
        lrow[row] = ltot;
        mrow[row] = mnew;
      }
      __syncthreads();
    }

    // ---- rescale accumulators, then P @ V (32 WMMAs per wave per block)
    float alpha_r[8];
#pragma unroll
    for (int r = 0; r < 8; ++r) alpha_r[r] = arow[r + half * 8];
#pragma unroll
    for (int nt = 0; nt < 8; ++nt)
#pragma unroll
      for (int r = 0; r < 8; ++r) accO[nt][r] *= alpha_r[r];

    const int vcolBase = wave * 128;
#pragma unroll
    for (int kk = 0; kk < 128; kk += 32) {
      FragBF pa;
      pa.q[0] = *(const uint4*)(&Ps[ln][kk + half * 16]);
      pa.q[1] = *(const uint4*)(&Ps[ln][kk + half * 16 + 8]);
#pragma unroll
      for (int nt = 0; nt < 8; ++nt) {
        FragBF vb;   // V^T: column = output dim, K(=key) contiguous
        const unsigned short* vp =
            Vb + (size_t)(vcolBase + nt * 16 + ln) * S_LEN + kBase + kk + half * 16;
        vb.q[0] = *(const uint4*)(vp);
        vb.q[1] = *(const uint4*)(vp + 8);
        accO[nt] = __builtin_amdgcn_wmma_f32_16x16x32_bf16(
            false, pa.v, false, vb.v, (short)0, accO[nt], false, false);
      }
    }
    __syncthreads();   // protect scoresS/Ps before next block
  }

  // ---- epilogue: divide by softmax denominator, store fp32
  float inv_l[8];
#pragma unroll
  for (int r = 0; r < 8; ++r) inv_l[r] = 1.0f / lrow[r + half * 8];
#pragma unroll
  for (int nt = 0; nt < 8; ++nt) {
    int gcol = wave * 128 + nt * 16 + ln;
#pragma unroll
    for (int r = 0; r < 8; ++r) {
      int grow = qBase + r + half * 8;
      out[((size_t)b * S_LEN + grow) * D_DIM + gcol] = accO[nt][r] * inv_l[r];
    }
  }
}

// ---------------------------------------------------------------------------
extern "C" void kernel_launch(void* const* d_in, const int* in_sizes, int n_in,
                              void* d_out, int out_size, void* d_ws, size_t ws_size,
                              hipStream_t stream) {
  (void)in_sizes; (void)n_in; (void)out_size; (void)ws_size;
  const float* x  = (const float*)d_in[0];
  const float* Wq = (const float*)d_in[1];
  const float* Wk = (const float*)d_in[2];
  const float* Wv = (const float*)d_in[3];

  const size_t elems = (size_t)BATCH * S_LEN * D_DIM;   // 8M
  unsigned short* Qw = (unsigned short*)d_ws;           // 16 MiB
  unsigned short* Kw = Qw + elems;                      // 16 MiB
  unsigned short* Vt = Kw + elems;                      // 16 MiB (transposed [B][D][S])
  float* out = (float*)d_out;

  qkv_proj_kernel<<<dim3(D_DIM / 128, (BATCH * S_LEN) / 128, 3), 256, 0, stream>>>(
      x, Wq, Wk, Wv, Qw, Kw, Vt);
  flash_attn_kernel<<<dim3((BATCH * S_LEN) / 16), 256, 0, stream>>>(Qw, Kw, Vt, out);
}
